// NestedLSTM_36189394436730
// MI455X (gfx1250) — compile-verified, matrix operand order checked by reading, not run on hardware
//
#include <hip/hip_runtime.h>
#include <hip/hip_bf16.h>
#include <math.h>

// Problem constants (from the reference)
#define GT 256    // time steps
#define GB 128    // batch
#define GC 512    // input channels
#define GH 512    // hidden
#define GK 1024   // GEMM K (C+H or 2H)
#define GN 2048   // 4*H gate columns
#define GBH (GB * GH)

typedef __attribute__((ext_vector_type(16))) _Float16 v16h;
typedef __attribute__((ext_vector_type(8)))  _Float16 v8h;
typedef __attribute__((ext_vector_type(8)))  float    v8f;

__device__ __forceinline__ float sigf(float x) { return 1.0f / (1.0f + expf(-x)); }

__device__ __forceinline__ v16h join8(v8h lo, v8h hi)
{
    v16h v;
#pragma unroll
    for (int i = 0; i < 8; ++i) { v[i] = lo[i]; v[i + 8] = hi[i]; }
    return v;
}

// ---------------------------------------------------------------------------
// GEMM: G[B, 4H] = A[B, K] * W[4H, K]^T + bias, f16 operands, f32 accum.
// A = concat(A0, A1) split statically at K=512 (two separate K half-loops,
// so addresses are pure induction -- no per-iteration operand selects).
// Block = 4 waves; each wave owns TWO adjacent 16x16 N tiles (A reuse x2).
// A staging: per K=128 chunk, wave w stages K-step w's 16x32 fragment into
// LDS, one barrier, then 4 K-steps x 2 tiles = 8 WMMAs, one barrier.
// ---------------------------------------------------------------------------
__global__ __launch_bounds__(128)
void gemm_gates_f16(const _Float16* __restrict__ A0, int lda0,
                    const _Float16* __restrict__ A1, int lda1,
                    const _Float16* __restrict__ W,     // [GN, GK] row-major f16
                    const float*    __restrict__ bias,  // [GN]
                    float*          __restrict__ G)     // [GB, GN] fp32
{
    __shared__ __attribute__((aligned(32))) _Float16 Ash[4 * 512]; // 4 fragments

    const int lane = threadIdx.x & 31;
    const int wave = threadIdx.x >> 5;
    const int row0 = blockIdx.x * 16;                    // M tile
    const int col0 = (blockIdx.y * 8 + wave * 2) * 16;   // first of 2 N tiles
    const int r    = lane & 15;
    const int grp  = (lane >> 4) * 8;                    // K sub-offset per half-wave

    v8f acc0 = {};
    v8f acc1 = {};

    // B-fragment row pointers for this wave's two N tiles (advance by k only)
    const _Float16* wp0 = W + (size_t)(col0 + r) * GK + grp;
    const _Float16* wp1 = W + (size_t)(col0 + 16 + r) * GK + grp;

#pragma unroll
    for (int half = 0; half < 2; ++half) {
        const _Float16* Ab = half ? A1 : A0;
        const int       ld = half ? lda1 : lda0;
        const _Float16* arow = Ab + (size_t)(row0 + r) * ld;

        for (int chunk = 0; chunk < 4; ++chunk) {            // K=128 per chunk
            const int kbase = half * 512 + chunk * 128;      // global K
            const int klocal = chunk * 128 + wave * 32;      // K within this half

            // Cooperative stage: wave w loads K-step w's fragment (2x16B/lane)
            {
                const _Float16* ap = arow + klocal + grp;
                v8h alo = *(const v8h*)ap;          // K = grp .. grp+7
                v8h ahi = *(const v8h*)(ap + 16);   // K = grp+16 .. grp+23
                *(v8h*)(Ash + wave * 512 + lane * 16)     = alo;
                *(v8h*)(Ash + wave * 512 + lane * 16 + 8) = ahi;
            }
            __syncthreads();

#pragma unroll
            for (int s = 0; s < 4; ++s) {
                v16h af = *(const v16h*)(Ash + s * 512 + lane * 16);
                const int k = kbase + s * 32;
                const _Float16* b0 = wp0 + k;
                const _Float16* b1 = wp1 + k;
                v8h b0lo = *(const v8h*)b0;
                v8h b0hi = *(const v8h*)(b0 + 16);
                v8h b1lo = *(const v8h*)b1;
                v8h b1hi = *(const v8h*)(b1 + 16);
                __builtin_prefetch(b0 + 128, 0, 1);   // next chunk of W rows
                v16h bf0 = join8(b0lo, b0hi);
                v16h bf1 = join8(b1lo, b1hi);
                acc0 = __builtin_amdgcn_wmma_f32_16x16x32_f16(
                           false, af, false, bf0, (short)0, acc0, false, false);
                acc1 = __builtin_amdgcn_wmma_f32_16x16x32_f16(
                           false, af, false, bf1, (short)0, acc1, false, false);
            }
            __syncthreads();
        }
    }

    // D layout: element rr -> (M = rr + 8*(lane>=16), N = lane&15)
    const int n  = lane & 15;
    const int mo = (lane >> 4) * 8;
    const float bv0 = bias[col0 + n];
    const float bv1 = bias[col0 + 16 + n];
#pragma unroll
    for (int rr = 0; rr < 8; ++rr) {
        const int m = rr + mo;
        float* grow = G + (size_t)(row0 + m) * GN;
        grow[col0 + n]      = acc0[rr] + bv0;
        grow[col0 + 16 + n] = acc1[rr] + bv1;
    }
}

// ---------------------------------------------------------------------------
// Elementwise: outer gates -> cell (f16, feeds inner GEMM as A0)
// ---------------------------------------------------------------------------
__global__ __launch_bounds__(256)
void outer_gates(const float* __restrict__ G, const float* __restrict__ c,
                 _Float16* __restrict__ cell16)
{
    const int idx = blockIdx.x * blockDim.x + threadIdx.x;   // < GB*GH
    const int b = idx >> 9;
    const int j = idx & (GH - 1);
    const float* g = G + (size_t)b * GN;
    const float iv  = g[j];
    const float fv  = g[GH + j];
    const float gcv = g[3 * GH + j];
    const float cell = sigf(fv) * c[idx] + sigf(iv) * tanhf(gcv);
    cell16[idx] = (_Float16)cell;
}

// ---------------------------------------------------------------------------
// Elementwise: inner gates -> state update + layer output
//   ci' = sig(f2)*ci + sig(i2)*tanh(g2c)
//   hi' = sig(o2)*tanh(ci');  c' = hi';  h' = sig(o)*tanh(hi')
// ---------------------------------------------------------------------------
__global__ __launch_bounds__(256)
void inner_gates(const float* __restrict__ G2, const float* __restrict__ G,
                 float* __restrict__ c, float* __restrict__ ci,
                 _Float16* __restrict__ h16, _Float16* __restrict__ hi16,
                 _Float16* __restrict__ y16, float* __restrict__ y32)
{
    const int idx = blockIdx.x * blockDim.x + threadIdx.x;   // < GB*GH
    const int b = idx >> 9;
    const int j = idx & (GH - 1);
    const float* g2 = G2 + (size_t)b * GN;
    const float i2  = g2[j];
    const float f2  = g2[GH + j];
    const float o2  = g2[2 * GH + j];
    const float g2c = g2[3 * GH + j];
    const float ov  = G[(size_t)b * GN + 2 * GH + j];

    const float cin = sigf(f2) * ci[idx] + sigf(i2) * tanhf(g2c);
    const float hin = sigf(o2) * tanhf(cin);
    const float hn  = sigf(ov) * tanhf(hin);

    ci[idx]  = cin;
    c[idx]   = hin;                 // outer cell state replaced by inner hidden
    hi16[idx] = (_Float16)hin;
    h16[idx]  = (_Float16)hn;
    if (y16) y16[idx] = (_Float16)hn;   // layer-0 output (f16, feeds layer 1)
    if (y32) y32[idx] = hn;             // layer-1 output (fp32, final result)
}

// ---------------------------------------------------------------------------
// fp32 -> f16 conversion (grid-stride)
// ---------------------------------------------------------------------------
__global__ __launch_bounds__(256)
void f32_to_f16(const float* __restrict__ src, _Float16* __restrict__ dst, int n)
{
    for (int i = blockIdx.x * blockDim.x + threadIdx.x; i < n;
         i += gridDim.x * blockDim.x)
        dst[i] = (_Float16)src[i];
}

// ---------------------------------------------------------------------------
// State init from h0[l, {h,c,hi,ci}, B, H]
// ---------------------------------------------------------------------------
__global__ __launch_bounds__(256)
void init_state(const float* __restrict__ h0, int l,
                _Float16* __restrict__ h16, float* __restrict__ c,
                _Float16* __restrict__ hi16, float* __restrict__ ci)
{
    const int idx = blockIdx.x * blockDim.x + threadIdx.x;
    if (idx >= GBH) return;
    const size_t BH = (size_t)GBH;
    const float* base = h0 + (size_t)l * 4 * BH;
    h16[idx]  = (_Float16)base[idx];
    c[idx]    = base[BH + idx];
    hi16[idx] = (_Float16)base[2 * BH + idx];
    ci[idx]   = base[3 * BH + idx];
}

extern "C" void kernel_launch(void* const* d_in, const int* in_sizes, int n_in,
                              void* d_out, int out_size, void* d_ws, size_t ws_size,
                              hipStream_t stream)
{
    (void)in_sizes; (void)n_in; (void)out_size; (void)ws_size;

    const float* x  = (const float*)d_in[0];   // (T, B, C)
    const float* h0 = (const float*)d_in[1];   // (L, 4, B, H)
    const float* Wg = (const float*)d_in[2];   // (L, 4H, C+H)
    const float* bg = (const float*)d_in[3];   // (L, 4H)
    const float* Wi = (const float*)d_in[4];   // (L, 4H, 2H)
    const float* bi = (const float*)d_in[5];   // (L, 4H)
    float* out = (float*)d_out;                // (T, B, H) fp32

    // Workspace carve-up (all offsets 256B aligned)
    char* ws = (char*)d_ws;
    _Float16* Wg16   = (_Float16*)(ws);                    //  8 MB: 2*2048*1024 f16
    _Float16* Wi16   = (_Float16*)(ws + 8388608);          //  8 MB
    _Float16* y016   = (_Float16*)(ws + 16777216);         // 32 MB: T*B*H f16 (layer-0 out)
    _Float16* xt16   = (_Float16*)(ws + 50331648);         // B*C f16 (per-step x)
    float*    G      = (float*)   (ws + 50462720);         // 1 MB: [B,4H]
    float*    G2     = (float*)   (ws + 51511296);         // 1 MB
    _Float16* cell16 = (_Float16*)(ws + 52559872);         // [B,H] f16
    _Float16* h16    = (_Float16*)(ws + 52690944);         // [B,H] f16
    _Float16* hi16   = (_Float16*)(ws + 52822016);         // [B,H] f16
    float*    c      = (float*)   (ws + 52953088);         // [B,H] fp32
    float*    ci     = (float*)   (ws + 53215232);         // [B,H] fp32

    // One-time weight conversion to f16 (weights then live in L2 across steps)
    f32_to_f16<<<2048, 256, 0, stream>>>(Wg, Wg16, 2 * GN * GK);
    f32_to_f16<<<2048, 256, 0, stream>>>(Wi, Wi16, 2 * GN * GK);

    const dim3 ggrid(GB / 16, GN / 128);  // 8 x 16 blocks, 128 threads each

    for (int l = 0; l < 2; ++l) {
        init_state<<<GBH / 256, 256, 0, stream>>>(h0, l, h16, c, hi16, ci);
        const _Float16* Wgl = Wg16 + (size_t)l * GN * GK;
        const _Float16* Wil = Wi16 + (size_t)l * GN * GK;
        const float* bgl = bg + (size_t)l * GN;
        const float* bil = bi + (size_t)l * GN;

        for (int t = 0; t < GT; ++t) {
            const _Float16* A0;
            if (l == 0) {
                f32_to_f16<<<GB * GC / 256, 256, 0, stream>>>(
                    x + (size_t)t * GB * GC, xt16, GB * GC);
                A0 = xt16;                                   // [B, 512], ld=512
            } else {
                A0 = y016 + (size_t)t * GBH;                 // [B, 512], ld=512
            }

            // Outer gates: G = concat(x_t, h) @ Wg^T + bg
            gemm_gates_f16<<<ggrid, 128, 0, stream>>>(A0, GC, h16, GH, Wgl, bgl, G);
            outer_gates<<<GBH / 256, 256, 0, stream>>>(G, c, cell16);

            // Inner gates: G2 = concat(cell, h_inner) @ Wi^T + bi
            gemm_gates_f16<<<ggrid, 128, 0, stream>>>(cell16, GH, hi16, GH, Wil, bil, G2);
            inner_gates<<<GBH / 256, 256, 0, stream>>>(
                G2, G, c, ci, h16, hi16,
                (l == 0) ? (y016 + (size_t)t * GBH) : (_Float16*)nullptr,
                (l == 1) ? (out + (size_t)t * GBH) : (float*)nullptr);
        }
    }
}
